// CFC_Reformer_61435212202311
// MI455X (gfx1250) — matrix operand order repaired and checked by reading, not compiled, not run on hardware
//
#include <hip/hip_runtime.h>
#include <hip/hip_bf16.h>

typedef __attribute__((ext_vector_type(16))) _Float16 v16h;
typedef __attribute__((ext_vector_type(8)))  _Float16 v8h;
typedef __attribute__((ext_vector_type(8)))  float    v8f;

#define BATCH 8
#define Hh    96
#define Ww    96
#define HWs   9216
#define NPOS  73728          // BATCH*Hh*Ww
#define CINc  256
#define COUTc 128
#define QDIM  32
#define NHEAD 8
#define LAMID 16
#define EPSBN 1e-5f

// ---- workspace layout (bytes, 256-aligned) ----
#define OFF_WPACK   0u            // 128*2304 f16 : conv1 weights, k = tap*256+cin
#define OFF_WPACK2  589824u       // 128*160  f16 : la2 weights, k = tap*16+cin (tap9 = 0)
#define OFF_WQH     630784u       // 32*128   f16
#define OFF_WLA1H   638976u       // 16*128   f16
#define OFF_KS      643072u       // 8*8*32   f32 : selected keys
#define OFF_VS      651264u       // 8*8*128  f32 : selected values
#define OFF_TOK     684032u       // 8*50*128 f32 : PSP tokens
#define OFF_XH      888832u       // NPOS*256 f16 : x NHWC f16
#define OFF_XR32    38637568u     // NPOS*128 f32 : xr NHWC
#define OFF_XRH     76386304u     // NPOS*128 f16 : xr NHWC
#define OFF_CTX32   95260672u     // NPOS*128 f32 : ctx NHWC
#define OFF_YH      133009408u    // NPOS*16  f16 : la1 output NHWC
// total = 135368704 bytes

static __device__ __forceinline__ v8f wmma16(v16h a, v16h b, v8f c) {
  return __builtin_amdgcn_wmma_f32_16x16x32_f16(false, a, false, b, (short)0, c,
                                                false, false);
}
static __device__ __forceinline__ v16h joinh(v8h lo, v8h hi) {
  union { v16h v; v8h h[2]; } u;
  u.h[0] = lo; u.h[1] = hi;
  return u.v;
}

// ------------------------------------------------------------------
// K0: weight prepack + f16 conversion
// ------------------------------------------------------------------
__global__ __launch_bounds__(256) void k_prepack(
    const float* __restrict__ wred, const float* __restrict__ wla2,
    const float* __restrict__ wq, const float* __restrict__ wla1,
    _Float16* __restrict__ Wp, _Float16* __restrict__ Wp2,
    _Float16* __restrict__ Wqh, _Float16* __restrict__ Wla1h) {
  int i = blockIdx.x * 256 + threadIdx.x;
  if (i < 128 * 2304) {
    int n = i / 2304, k = i % 2304, tap = k / CINc, cin = k % CINc;
    Wp[i] = (_Float16)wred[((size_t)n * CINc + cin) * 9 + tap];
    return;
  }
  i -= 128 * 2304;
  if (i < 128 * 160) {
    int n = i / 160, k = i % 160, tap = k / 16, cin = k % 16;
    Wp2[i] = (_Float16)((tap < 9) ? wla2[((size_t)n * 16 + cin) * 9 + tap] : 0.f);
    return;
  }
  i -= 128 * 160;
  if (i < 32 * 128) { Wqh[i] = (_Float16)wq[i]; return; }
  i -= 32 * 128;
  if (i < 16 * 128) { Wla1h[i] = (_Float16)wla1[i]; }
}

// ------------------------------------------------------------------
// K0b: x NCHW fp32 -> xh NHWC f16 (LDS tile transpose, 64c x 64hw per block)
// ------------------------------------------------------------------
__global__ __launch_bounds__(256) void k_xhalf(const float* __restrict__ x,
                                               _Float16* __restrict__ xh) {
  __shared__ _Float16 tile[64 * 72];  // [hw][c], padded stride
  const int t = threadIdx.x;
  const int hb = blockIdx.x;          // 144 tiles of 64 positions
  const int cb = blockIdx.y;          // 4 tiles of 64 channels
  const int bb = blockIdx.z;          // 8 batches
  const int c0 = cb * 64, hw0 = hb * 64;
  const float* src = x + ((size_t)bb * CINc + c0) * HWs + hw0;
#pragma unroll
  for (int it = 0; it < 4; ++it) {
    const int unit = it * 256 + t;     // 1024 float4 units
    const int cc = unit / 16;          // channel row
    const int w4 = (unit % 16) * 4;    // hw within tile
    const float4 v = *(const float4*)(src + (size_t)cc * HWs + w4);
    tile[(w4 + 0) * 72 + cc] = (_Float16)v.x;
    tile[(w4 + 1) * 72 + cc] = (_Float16)v.y;
    tile[(w4 + 2) * 72 + cc] = (_Float16)v.z;
    tile[(w4 + 3) * 72 + cc] = (_Float16)v.w;
  }
  __syncthreads();
  _Float16* dst = xh + ((size_t)bb * HWs + hw0) * CINc + c0;
#pragma unroll
  for (int it = 0; it < 2; ++it) {
    const int unit = it * 256 + t;     // 512 v8h units
    const int pp = unit / 8, gi = unit % 8;
    const v8h vv = *(const v8h*)&tile[pp * 72 + gi * 8];
    *(v8h*)(dst + (size_t)pp * CINc + gi * 8) = vv;
  }
}

// ------------------------------------------------------------------
// K1: 3x3 conv 256->128 (implicit GEMM, f16 WMMA, barrier-free) + BN + SiLU
// A fragments loaded directly from xh (NHWC f16) as contiguous b128 chunks.
// B fragments preloaded per K-step into a register array so the 16 weight
// loads issue as one clause and overlap the 8-WMMA chain.
// block = 128 thr (4 waves), tile = 64 positions x 128 couts
// ------------------------------------------------------------------
__global__ __launch_bounds__(128) void k_conv_red(
    const _Float16* __restrict__ xh, const _Float16* __restrict__ Wp,
    const float* __restrict__ bng, const float* __restrict__ bnb,
    const float* __restrict__ bnm, const float* __restrict__ bnv,
    float* __restrict__ xr32, _Float16* __restrict__ xrh) {
  const int t = threadIdx.x;
  const int p0 = blockIdx.x * 64;
  const int lane = t & 31, wv = t >> 5, hi = lane >> 4, lr = lane & 15;
  const int p = p0 + wv * 16 + lr;
  const int b = p / HWs, hw = p % HWs, h = hw / Ww, w = hw % Ww;
  const _Float16* wrow = Wp + (size_t)lr * 2304 + 8 * hi;
  v8f acc[8] = {};
  for (int tap = 0; tap < 9; ++tap) {
    const int ih = h + tap / 3 - 1, iw = w + tap % 3 - 1;
    const bool valid = ((unsigned)ih < Hh) && ((unsigned)iw < Ww);
    const _Float16* ap =
        xh + ((size_t)b * HWs + (valid ? (ih * Ww + iw) : 0)) * CINc + 8 * hi;
    const _Float16* wk = wrow + tap * CINc;
#pragma unroll
    for (int cb = 0; cb < 8; ++cb) {
      v16h a = {};
      if (valid)
        a = joinh(*(const v8h*)(ap + cb * 32), *(const v8h*)(ap + cb * 32 + 16));
      v16h bf[8];
#pragma unroll
      for (int nb = 0; nb < 8; ++nb)
        bf[nb] = joinh(*(const v8h*)(wk + cb * 32 + (size_t)nb * 16 * 2304),
                       *(const v8h*)(wk + cb * 32 + 16 + (size_t)nb * 16 * 2304));
#pragma unroll
      for (int nb = 0; nb < 8; ++nb) acc[nb] = wmma16(a, bf[nb], acc[nb]);
    }
  }
#pragma unroll
  for (int nb = 0; nb < 8; ++nb) {
    const int n = nb * 16 + lr;
    const float inv = bng[n] * rsqrtf(bnv[n] + EPSBN);
    const float beta = bnb[n] - bnm[n] * inv;
#pragma unroll
    for (int r = 0; r < 8; ++r) {
      const size_t pp = (size_t)(p0 + wv * 16 + r + 8 * hi);
      const float y = acc[nb][r] * inv + beta;
      const float s = y / (1.f + __expf(-y));  // SiLU
      xr32[pp * COUTc + n] = s;
      xrh[pp * COUTc + n] = (_Float16)s;
    }
  }
}

// ------------------------------------------------------------------
// K2: PSP adaptive-avg-pool tokens [8][50][128]
// ------------------------------------------------------------------
__global__ __launch_bounds__(128) void k_tokens(const float* __restrict__ xr32,
                                                float* __restrict__ tok) {
  const int b = blockIdx.x / 50, tk = blockIdx.x % 50;
  const int c = threadIdx.x;
  int h0, h1, w0, w1;
  if (tk < 36) { int i = tk / 6, j = tk % 6; h0 = i * 16; h1 = h0 + 16; w0 = j * 16; w1 = w0 + 16; }
  else if (tk < 45) { int u = tk - 36, i = u / 3, j = u % 3; h0 = i * 32; h1 = h0 + 32; w0 = j * 32; w1 = w0 + 32; }
  else if (tk < 49) { int u = tk - 45, i = u / 2, j = u % 2; h0 = i * 48; h1 = h0 + 48; w0 = j * 48; w1 = w0 + 48; }
  else { h0 = 0; h1 = 96; w0 = 0; w1 = 96; }
  float s = 0.f;
  for (int h = h0; h < h1; ++h) {
    const float* row = xr32 + ((size_t)b * HWs + h * Ww + w0) * COUTc + c;
    for (int w = 0; w < (w1 - w0); ++w) s += row[(size_t)w * COUTc];
  }
  tok[((size_t)b * 50 + tk) * COUTc + c] = s / (float)((h1 - h0) * (w1 - w0));
}

// ------------------------------------------------------------------
// K3: k_tok for all 50 tokens, LSH argmax buckets, stable argsort,
//     gather k_s [8][8][32], compute v_s [8][8][128]. One block per batch.
// (Softmax attention is invariant under the joint k/v permutation, but we
//  implement the argmax/argsort faithfully anyway - it is negligible work.)
// ------------------------------------------------------------------
__global__ __launch_bounds__(128) void k_kv_sort(
    const float* __restrict__ tok, const float* __restrict__ wk,
    const float* __restrict__ bk, const float* __restrict__ wv,
    const float* __restrict__ bv, const float* __restrict__ lsh,
    float* __restrict__ ks_o, float* __restrict__ vs_o) {
  __shared__ float ktok[50 * 32];
  __shared__ float sc[NHEAD * 50];
  __shared__ int kidx[NHEAD];
  const int b = blockIdx.x, t = threadIdx.x;
  const float* tb = tok + (size_t)b * 50 * COUTc;
  for (int u = t; u < 50 * 32; u += 128) {
    const int n = u / 32, d = u % 32;
    float s = bk[d];
    const float* wr = wk + (size_t)d * COUTc;
    const float* tr = tb + (size_t)n * COUTc;
    for (int c = 0; c < COUTc; ++c) s += wr[c] * tr[c];
    ktok[u] = s;
  }
  __syncthreads();
  for (int u = t; u < NHEAD * 50; u += 128) {
    const int hh = u / 50, n = u % 50;
    float s = 0.f;
    const float* lr = lsh + hh * 32;
    const float* kr = ktok + n * 32;
    for (int d = 0; d < 32; ++d) s += lr[d] * kr[d];
    sc[u] = s;
  }
  __syncthreads();
  if (t == 0) {
    int bkt[NHEAD];
    for (int hh = 0; hh < NHEAD; ++hh) {
      int best = 0; float bvv = sc[hh * 50];
      for (int n = 1; n < 50; ++n) {
        const float v = sc[hh * 50 + n];
        if (v > bvv) { bvv = v; best = n; }
      }
      bkt[hh] = best;
    }
    int ord[NHEAD];
    for (int i = 0; i < NHEAD; ++i) ord[i] = i;
    for (int i = 1; i < NHEAD; ++i) {  // stable insertion sort (ascending)
      int j = i;
      while (j > 0 && bkt[ord[j - 1]] > bkt[ord[j]]) {
        const int tmp = ord[j - 1]; ord[j - 1] = ord[j]; ord[j] = tmp; --j;
      }
    }
    for (int i = 0; i < NHEAD; ++i) kidx[i] = ord[i];
  }
  __syncthreads();
  for (int u = t; u < NHEAD * 32; u += 128) {
    const int j = u / 32, d = u % 32;
    ks_o[((size_t)b * NHEAD + j) * 32 + d] = ktok[kidx[j] * 32 + d];
  }
  for (int u = t; u < NHEAD * COUTc; u += 128) {
    const int j = u / COUTc, c = u % COUTc, n = kidx[j];
    float s = bv[c];
    const float* wr = wv + (size_t)c * COUTc;
    const float* tr = tb + (size_t)n * COUTc;
    for (int c2 = 0; c2 < COUTc; ++c2) s += wr[c2] * tr[c2];
    vs_o[((size_t)b * NHEAD + j) * COUTc + c] = s;
  }
}

// ------------------------------------------------------------------
// K4: fused q-proj (WMMA) + 8-key softmax attention + ctx + la1 (WMMA)
// block = 128 thr (4 waves), tile = 64 positions
// ------------------------------------------------------------------
__global__ __launch_bounds__(128) void k_attn(
    const _Float16* __restrict__ xrh, const _Float16* __restrict__ Wqh,
    const float* __restrict__ bq, const float* __restrict__ ks_i,
    const float* __restrict__ vs_i, const _Float16* __restrict__ Wla1h,
    const float* __restrict__ lag, const float* __restrict__ lab,
    const float* __restrict__ lam, const float* __restrict__ lav,
    float* __restrict__ ctx32, _Float16* __restrict__ yh) {
  __shared__ float qs[64 * 33];
  __shared__ float att[64 * NHEAD];
  __shared__ _Float16 ct[64 * 136];  // ctx tile f16, padded stride
  const int t = threadIdx.x;
  const int p0 = blockIdx.x * 64;
  const int bidx = p0 / HWs;
  const int lane = t & 31, wv = t >> 5, hi = lane >> 4, lr = lane & 15;
  // ---- q = Wq . xr + bq (WMMA, M=64, N=32, K=128) ----
  v8f q0 = {}, q1 = {};
  const _Float16* arow = xrh + (size_t)(p0 + wv * 16 + lr) * COUTc;
#pragma unroll
  for (int ks = 0; ks < 4; ++ks) {
    const int kb = ks * 32;
    const v16h a = joinh(*(const v8h*)(arow + kb + 8 * hi),
                         *(const v8h*)(arow + kb + 16 + 8 * hi));
    const v16h b0 = joinh(*(const v8h*)(Wqh + (size_t)lr * COUTc + kb + 8 * hi),
                          *(const v8h*)(Wqh + (size_t)lr * COUTc + kb + 16 + 8 * hi));
    const v16h b1 = joinh(*(const v8h*)(Wqh + (size_t)(16 + lr) * COUTc + kb + 8 * hi),
                          *(const v8h*)(Wqh + (size_t)(16 + lr) * COUTc + kb + 16 + 8 * hi));
    q0 = wmma16(a, b0, q0);
    q1 = wmma16(a, b1, q1);
  }
  {
    const float bq0 = bq[lr], bq1 = bq[16 + lr];
#pragma unroll
    for (int r = 0; r < 8; ++r) {
      const int mm = wv * 16 + r + 8 * hi;
      qs[mm * 33 + lr] = q0[r] + bq0;
      qs[mm * 33 + 16 + lr] = q1[r] + bq1;
    }
  }
  __syncthreads();
  // ---- scores + softmax over 8 keys (one thread per position) ----
  if (t < 64) {
    const float* qrow = &qs[t * 33];
    float s[NHEAD], mx = -3.4e38f;
    for (int k = 0; k < NHEAD; ++k) {
      const float* kr = ks_i + ((size_t)bidx * NHEAD + k) * 32;
      float v = 0.f;
      for (int d = 0; d < 32; ++d) v += qrow[d] * kr[d];
      s[k] = v; mx = fmaxf(mx, v);
    }
    float sum = 0.f;
    for (int k = 0; k < NHEAD; ++k) { s[k] = __expf(s[k] - mx); sum += s[k]; }
    const float rs = 1.f / sum;
    for (int k = 0; k < NHEAD; ++k) att[t * NHEAD + k] = s[k] * rs;
  }
  __syncthreads();
  // ---- ctx = att @ v_s (thread = channel) ----
  {
    const int c = t;
    float vv[NHEAD];
#pragma unroll
    for (int k = 0; k < NHEAD; ++k)
      vv[k] = vs_i[((size_t)bidx * NHEAD + k) * COUTc + c];
    for (int m = 0; m < 64; ++m) {
      float v = 0.f;
#pragma unroll
      for (int k = 0; k < NHEAD; ++k) v += att[m * NHEAD + k] * vv[k];
      ctx32[(size_t)(p0 + m) * COUTc + c] = v;
      ct[m * 136 + c] = (_Float16)v;
    }
  }
  __syncthreads();
  // ---- la1: 1x1 conv 128->16 + BN + SiLU (WMMA, N=16, K=128) ----
  v8f yacc = {};
  const _Float16* crow = &ct[(wv * 16 + lr) * 136];
#pragma unroll
  for (int ks = 0; ks < 4; ++ks) {
    const int kb = ks * 32;
    const v16h a = joinh(*(const v8h*)(crow + kb + 8 * hi),
                         *(const v8h*)(crow + kb + 16 + 8 * hi));
    const v16h bf = joinh(*(const v8h*)(Wla1h + (size_t)lr * COUTc + kb + 8 * hi),
                          *(const v8h*)(Wla1h + (size_t)lr * COUTc + kb + 16 + 8 * hi));
    yacc = wmma16(a, bf, yacc);
  }
  {
    const int n = lr;
    const float inv = lag[n] * rsqrtf(lav[n] + EPSBN);
    const float beta = lab[n] - lam[n] * inv;
#pragma unroll
    for (int r = 0; r < 8; ++r) {
      const size_t pp = (size_t)(p0 + wv * 16 + r + 8 * hi);
      const float y = yacc[r] * inv + beta;
      const float sv = y / (1.f + __expf(-y));
      yh[pp * 16 + n] = (_Float16)sv;
    }
  }
}

// ------------------------------------------------------------------
// K5: 3x3 conv 16->128 (implicit GEMM, K padded 144->160, barrier-free)
//     + tanh mask + residual combine, NHWC -> NCHW final store.
// A fragment: chunk0 = tap 2s (8 contiguous halves), chunk1 = tap 2s+1.
// B fragments preloaded per K-step (register array) for load/WMMA overlap.
// ------------------------------------------------------------------
__global__ __launch_bounds__(128) void k_la2_final(
    const _Float16* __restrict__ yh, const _Float16* __restrict__ Wp2,
    const float* __restrict__ ctx32, const float* __restrict__ xr32,
    float* __restrict__ out) {
  const int t = threadIdx.x;
  const int p0 = blockIdx.x * 64;
  const int lane = t & 31, wv = t >> 5, hi = lane >> 4, lr = lane & 15;
  const int p = p0 + wv * 16 + lr;
  const int b = p / HWs, hw = p % HWs, h = hw / Ww, w = hw % Ww;
  const _Float16* yb = yh + (size_t)b * HWs * 16 + 8 * hi;
  const _Float16* w2row = Wp2 + (size_t)lr * 160 + 8 * hi;
  v8f acc[8] = {};
#pragma unroll
  for (int s = 0; s < 5; ++s) {  // K-step = taps {2s, 2s+1}; tap 9 == zeros
    const int t0 = 2 * s, t1 = 2 * s + 1;
    v8h c0 = {}, c1 = {};
    {
      const int ih = h + t0 / 3 - 1, iw = w + t0 % 3 - 1;
      if (((unsigned)ih < Hh) && ((unsigned)iw < Ww))
        c0 = *(const v8h*)(yb + (size_t)(ih * Ww + iw) * 16);
    }
    if (t1 < 9) {
      const int ih = h + t1 / 3 - 1, iw = w + t1 % 3 - 1;
      if (((unsigned)ih < Hh) && ((unsigned)iw < Ww))
        c1 = *(const v8h*)(yb + (size_t)(ih * Ww + iw) * 16);
    }
    const v16h a = joinh(c0, c1);
    v16h bf[8];
#pragma unroll
    for (int nb = 0; nb < 8; ++nb)
      bf[nb] = joinh(*(const v8h*)(w2row + s * 32 + (size_t)nb * 16 * 160),
                     *(const v8h*)(w2row + s * 32 + 16 + (size_t)nb * 16 * 160));
#pragma unroll
    for (int nb = 0; nb < 8; ++nb) acc[nb] = wmma16(a, bf[nb], acc[nb]);
  }
  // out = xr + ctx * (1 + tanh(conv));  NHWC reads, NCHW 32B-contiguous stores
  const int bidx = p0 / HWs;
  const size_t pBase = (size_t)(p0 + wv * 16 + 8 * hi);
  const int hw0 = (int)(pBase % HWs);
#pragma unroll
  for (int nb = 0; nb < 8; ++nb) {
    const int n = nb * 16 + lr;
    float o[8];
#pragma unroll
    for (int r = 0; r < 8; ++r) {
      const size_t pp = pBase + r;
      const float cv = ctx32[pp * COUTc + n];
      const float xv = xr32[pp * COUTc + n];
      const float mk = tanhf(acc[nb][r]);
      o[r] = xv + cv * (1.f + mk);
    }
    float* op = out + ((size_t)bidx * COUTc + n) * HWs + hw0;
    *(float4*)(op) = make_float4(o[0], o[1], o[2], o[3]);
    *(float4*)(op + 4) = make_float4(o[4], o[5], o[6], o[7]);
  }
}

// ------------------------------------------------------------------
extern "C" void kernel_launch(void* const* d_in, const int* in_sizes, int n_in,
                              void* d_out, int out_size, void* d_ws,
                              size_t ws_size, hipStream_t stream) {
  (void)in_sizes; (void)n_in; (void)out_size; (void)ws_size;
  const float* x     = (const float*)d_in[0];
  const float* w_red = (const float*)d_in[1];
  const float* bng   = (const float*)d_in[2];
  const float* bnb   = (const float*)d_in[3];
  const float* bnm   = (const float*)d_in[4];
  const float* bnv   = (const float*)d_in[5];
  const float* w_q   = (const float*)d_in[6];
  const float* b_q   = (const float*)d_in[7];
  const float* w_k   = (const float*)d_in[8];
  const float* b_k   = (const float*)d_in[9];
  const float* w_v   = (const float*)d_in[10];
  const float* b_v   = (const float*)d_in[11];
  const float* lsh   = (const float*)d_in[12];
  const float* w_la1 = (const float*)d_in[13];
  const float* lag   = (const float*)d_in[14];
  const float* lab   = (const float*)d_in[15];
  const float* lam   = (const float*)d_in[16];
  const float* lav   = (const float*)d_in[17];
  const float* w_la2 = (const float*)d_in[18];
  float* out = (float*)d_out;

  char* ws = (char*)d_ws;
  _Float16* Wp    = (_Float16*)(ws + OFF_WPACK);
  _Float16* Wp2   = (_Float16*)(ws + OFF_WPACK2);
  _Float16* Wqh   = (_Float16*)(ws + OFF_WQH);
  _Float16* Wla1h = (_Float16*)(ws + OFF_WLA1H);
  float*    ksb   = (float*)(ws + OFF_KS);
  float*    vsb   = (float*)(ws + OFF_VS);
  float*    tokb  = (float*)(ws + OFF_TOK);
  _Float16* xhb   = (_Float16*)(ws + OFF_XH);
  float*    xr32  = (float*)(ws + OFF_XR32);
  _Float16* xrh   = (_Float16*)(ws + OFF_XRH);
  float*    ctx32 = (float*)(ws + OFF_CTX32);
  _Float16* yhb   = (_Float16*)(ws + OFF_YH);

  const int nTiles = NPOS / 64;  // 1152

  k_prepack<<<1256, 256, 0, stream>>>(w_red, w_la2, w_q, w_la1, Wp, Wp2, Wqh, Wla1h);
  k_xhalf<<<dim3(144, 4, BATCH), 256, 0, stream>>>(x, xhb);
  k_conv_red<<<nTiles, 128, 0, stream>>>(xhb, Wp, bng, bnb, bnm, bnv, xr32, xrh);
  k_tokens<<<BATCH * 50, 128, 0, stream>>>(xr32, tokb);
  k_kv_sort<<<BATCH, 128, 0, stream>>>(tokb, w_k, b_k, w_v, b_v, lsh, ksb, vsb);
  k_attn<<<nTiles, 128, 0, stream>>>(xrh, Wqh, b_q, ksb, vsb, Wla1h, lag, lab,
                                     lam, lav, ctx32, yhb);
  k_la2_final<<<nTiles, 128, 0, stream>>>(yhb, Wp2, ctx32, xr32, out);
}
